// sb_vq_layer_5128190951934
// MI455X (gfx1250) — matrix operand order repaired and checked by reading, not compiled, not run on hardware
//
#include <hip/hip_runtime.h>
#include <hip/hip_bf16.h>

typedef __bf16 bf16_t;
typedef __attribute__((ext_vector_type(8)))  bf16_t v8bf;
typedef __attribute__((ext_vector_type(16))) bf16_t v16bf;
typedef __attribute__((ext_vector_type(8)))  float  v8f;
typedef __attribute__((ext_vector_type(4)))  float  v4f;
typedef __attribute__((ext_vector_type(4)))  unsigned int u32x4;
typedef __attribute__((ext_vector_type(8)))  int i32x8;
typedef __attribute__((ext_vector_type(4)))  int i32x4;

#define N_TOKENS 65536
#define KLAT     1024
#define DIM      256
#define MT       32     // latent rows per workgroup
#define NCHUNK   128    // codebook entries per LDS chunk
#define LDA      264    // bf16 element stride for A/B tiles (bank-conflict pad: +16B/row)
#define LDD      1028   // f32 element stride for distance tile

#ifdef __has_builtin
#  if __has_builtin(__builtin_amdgcn_tensor_load_to_lds)
#    define HAVE_TDM 1
#  endif
#endif
#ifndef HAVE_TDM
#  define HAVE_TDM 0
#endif

// Issue one TDM 2D tile load: 128 rows x 512B (64 x 8B elements) of bf16 codebook
// into LDS, with 16B padding inserted after every 512B row (matches LDA=264 stride).
__device__ __forceinline__ void tdm_issue_chunk(const bf16_t* gsrc, unsigned lds_off) {
#if HAVE_TDM
  unsigned long long ga = (unsigned long long)(size_t)gsrc;
  u32x4 g0;
  g0[0] = 1u;                                              // count=1 user descriptor
  g0[1] = lds_off;                                         // LDS byte address of tile
  g0[2] = (unsigned)ga;                                    // global_addr[31:0]
  g0[3] = (unsigned)((ga >> 32) & 0x1FFFFFFull)            // global_addr[56:32]
        | (2u << 30);                                      // type = 2 ("image")
  i32x8 g1;
  g1[0] = (int)((3u << 16)      // data_size = 8 bytes
        |       (1u << 20)      // pad_enable
        |       (6u << 22)      // pad_interval: 2^(6+1)=128 DWORDs (512B) between pads
        |       (3u << 25));    // pad_amount: 4 DWORDs (16B)
  g1[1] = (int)(64u << 16);     // tensor_dim0 = 64 elements (8B units)
  g1[2] = (int)(1024u << 16);   // tensor_dim1[15:0] = 1024 rows
  g1[3] = (int)(64u << 16);     // tile_dim0 = 64 elements
  g1[4] = (int)128u;            // tile_dim1 = 128 rows, tile_dim2 = 0
  g1[5] = 64;                   // tensor_dim0_stride = 64 elements
  g1[6] = 0;
  g1[7] = 0;
  i32x4 g2 = {0, 0, 0, 0};
  i32x4 g3 = {0, 0, 0, 0};
#if __clang_major__ >= 23
  i32x8 g4 = {0, 0, 0, 0, 0, 0, 0, 0};
  __builtin_amdgcn_tensor_load_to_lds(g0, g1, g2, g3, g4, 0);
#else
  __builtin_amdgcn_tensor_load_to_lds(g0, g1, g2, g3, 0);
#endif
#else
  (void)gsrc; (void)lds_off;
#endif
}

// ---------------- prep: bf16 codebook + ||c_k||^2 + zero accumulators ----------------
__global__ __launch_bounds__(256) void vq_prep(const float* __restrict__ cb,
                                               bf16_t* __restrict__ cb_bf,
                                               float* __restrict__ cnorm,
                                               float* __restrict__ probs_acc,
                                               float* __restrict__ loss_acc) {
  int row = blockIdx.x;          // one codebook row per block (1024 blocks)
  int t   = threadIdx.x;         // 256 threads == DIM
  float v = cb[(size_t)row * DIM + t];
  cb_bf[(size_t)row * DIM + t] = (bf16_t)v;
  float s = v * v;
  for (int off = 16; off; off >>= 1) s += __shfl_xor(s, off);
  __shared__ float red[8];
  int wave = t >> 5, lane = t & 31;
  if (lane == 0) red[wave] = s;
  __syncthreads();
  if (t == 0) {
    float tot = 0.f;
    for (int i = 0; i < 8; ++i) tot += red[i];
    cnorm[row] = tot;
  }
  if (row < 4) probs_acc[row * 256 + t] = 0.f;
  if (row == 0 && t == 0) loss_acc[0] = 0.f;
}

// ---------------- main: fused distance GEMM + argmin + softmax + gather ----------------
__global__ __launch_bounds__(256) void vq_main(const float*  __restrict__ z,
                                               const float*  __restrict__ cb,
                                               const bf16_t* __restrict__ cb_bf,
                                               const float*  __restrict__ cnorm,
                                               float* __restrict__ qout,
                                               float* __restrict__ probs_acc,
                                               float* __restrict__ loss_acc) {
  extern __shared__ char smem[];
  bf16_t* As  = (bf16_t*)smem;                                  // [MT][LDA] bf16
  bf16_t* Bs0 = (bf16_t*)(smem + MT * LDA * 2);                 // [NCHUNK][LDA] bf16
  bf16_t* Bs1 = Bs0 + NCHUNK * LDA;                             // [NCHUNK][LDA] bf16
  float*  Ds  = (float*)(smem + (MT + 2 * NCHUNK) * LDA * 2);   // [MT][LDD] f32
  float* pAcc = (float*)((char*)Ds + MT * LDD * 4);             // [KLAT]
  float* lAcc = pAcc + KLAT;                                    // [1]

  const int t    = threadIdx.x;
  const int wave = t >> 5;
  const int lane = t & 31;
  const int half = lane >> 4;
  const int l16  = lane & 15;
  const int rowBase = blockIdx.x * MT;

  for (int k = t; k < KLAT; k += 256) pAcc[k] = 0.f;
  if (t == 0) lAcc[0] = 0.f;

  // Load Z tile (32x256 f32) -> bf16 LDS, 128-bit global loads
  {
    const v4f* zg = (const v4f*)(z + (size_t)rowBase * DIM);
    for (int it = 0; it < 8; ++it) {
      int c  = it * 256 + t;           // float4 chunk id, 0..2047
      int r  = c >> 6;                 // row 0..31
      int c4 = c & 63;
      v4f v = zg[c];
      bf16_t* dst = As + r * LDA + c4 * 4;
      dst[0] = (bf16_t)v[0]; dst[1] = (bf16_t)v[1];
      dst[2] = (bf16_t)v[2]; dst[3] = (bf16_t)v[3];
    }
  }

#if HAVE_TDM
  const unsigned bsoff0 = (unsigned)(size_t)(void*)Bs0;
  const unsigned bsoff1 = (unsigned)(size_t)(void*)Bs1;
  if (wave == 0) tdm_issue_chunk(cb_bf, bsoff0);        // prefetch chunk 0 -> buf0
#endif

  // Tile assignment: wave w -> m-block (w&1), n-blocks 2*(w>>1), 2*(w>>1)+1
  const int mblk = wave & 1;
  const int nb0  = (wave >> 1) * 2;
  const int am   = mblk * 16 + l16;    // A row for this lane (ISA A-frag layout)

  for (int nc = 0; nc < KLAT / NCHUNK; ++nc) {
    bf16_t* Bs = (nc & 1) ? Bs1 : Bs0;
#if HAVE_TDM
    if (wave == 0) {
      if (nc < KLAT / NCHUNK - 1) {
        // prefetch next chunk into the other buffer (freed by end-barrier of nc-1)
        tdm_issue_chunk(cb_bf + (size_t)(nc + 1) * NCHUNK * DIM,
                        (nc & 1) ? bsoff0 : bsoff1);
        __builtin_amdgcn_s_wait_tensorcnt(1);   // chunk nc landed, nc+1 in flight
      } else {
        __builtin_amdgcn_s_wait_tensorcnt(0);   // last chunk landed
      }
    }
    __syncthreads();                   // TDM data visible + As ready (1st iter)
#else
    __syncthreads();                   // Bs reuse hazard (also covers As writes)
    {                                  // fallback: manual 128x256 bf16 chunk copy
      const v8bf* bg = (const v8bf*)(cb_bf + (size_t)nc * NCHUNK * DIM);
      for (int it = 0; it < 16; ++it) {
        int c = it * 256 + t;          // 8-elt chunk id, 0..4095
        int r = c >> 5;
        int o = (c & 31) * 8;
        *(v8bf*)(Bs + r * LDA + o) = bg[c];
      }
    }
    __syncthreads();
#endif

    v8f acc0 = {}; v8f acc1 = {};
    const int n0 = nb0 * 16 + l16;     // B rows (codebook entries) for the two tiles
    const int n1 = n0 + 16;
    for (int kc = 0; kc < DIM; kc += 32) {
      // A fragment: lane holds row `am`, K run [half*16,+16)
      v8bf alo = *(const v8bf*)(As + am * LDA + kc + half * 16);
      v8bf ahi = *(const v8bf*)(As + am * LDA + kc + half * 16 + 8);
      v16bf a = __builtin_shufflevector(alo, ahi, 0,1,2,3,4,5,6,7,8,9,10,11,12,13,14,15);
      // B fragment: lane holds column n, K run [half*16,+16)
      v8bf b0lo = *(const v8bf*)(Bs + n0 * LDA + kc + half * 16);
      v8bf b0hi = *(const v8bf*)(Bs + n0 * LDA + kc + half * 16 + 8);
      v16bf b0 = __builtin_shufflevector(b0lo, b0hi, 0,1,2,3,4,5,6,7,8,9,10,11,12,13,14,15);
      acc0 = __builtin_amdgcn_wmma_f32_16x16x32_bf16(false, a, false, b0, (short)0, acc0, false, false);
      v8bf b1lo = *(const v8bf*)(Bs + n1 * LDA + kc + half * 16);
      v8bf b1hi = *(const v8bf*)(Bs + n1 * LDA + kc + half * 16 + 8);
      v16bf b1 = __builtin_shufflevector(b1lo, b1hi, 0,1,2,3,4,5,6,7,8,9,10,11,12,13,14,15);
      acc1 = __builtin_amdgcn_wmma_f32_16x16x32_bf16(false, a, false, b1, (short)0, acc1, false, false);
    }
    // e[m,k] = ||c_k||^2 - 2*dot  (||z||^2 cancels in argmin & softmax)
    int ng0 = nc * NCHUNK + nb0 * 16 + l16;
    float cn0 = cnorm[ng0];
    float cn1 = cnorm[ng0 + 16];
    int cbase = nc * NCHUNK + nb0 * 16;
    for (int r = 0; r < 8; ++r) {
      int m = mblk * 16 + half * 8 + r;     // ISA C/D layout: vgpr r, lane-half
      Ds[m * LDD + cbase + l16]      = cn0 - 2.f * acc0[r];
      Ds[m * LDD + cbase + 16 + l16] = cn1 - 2.f * acc1[r];
    }
    __syncthreads();                   // all reads of Bs done before it is refilled
  }

  // Per-row: argmin, softmax accumulation, quantized gather + loss (4 rows per wave)
  for (int rr = 0; rr < 4; ++rr) {
    int m = wave * 4 + rr;
    const float* drow = Ds + m * LDD;
    float mv = 3.4e38f; int mi = 0;
    for (int j = 0; j < KLAT / 32; ++j) {
      int k = j * 32 + lane;
      float v = drow[k];
      if (v < mv) { mv = v; mi = k; }
    }
    for (int off = 16; off; off >>= 1) {
      float ov = __shfl_xor(mv, off);
      int   oi = __shfl_xor(mi, off);
      if (ov < mv || (ov == mv && oi < mi)) { mv = ov; mi = oi; }
    }
    float s = 0.f;
    for (int j = 0; j < KLAT / 32; ++j)
      s += __expf(mv - drow[j * 32 + lane]);          // max(-e) = -min(e): terms <= 1
    for (int off = 16; off; off >>= 1) s += __shfl_xor(s, off);
    float inv = 1.f / s;
    for (int j = 0; j < KLAT / 32; ++j) {
      int k = j * 32 + lane;
      atomicAdd(&pAcc[k], __expf(mv - drow[k]) * inv); // ds_add_f32, conflict-free lanes
    }
    // quantized row copy (exact f32 codebook) + loss vs exact f32 z
    int zr = rowBase + m;
    const v4f* cbr = (const v4f*)(cb + (size_t)mi * DIM);
    const v4f* zr4 = (const v4f*)(z + (size_t)zr * DIM);
    v4f* q4 = (v4f*)(qout + (size_t)zr * DIM);
    float le = 0.f;
    for (int j = 0; j < 2; ++j) {
      int c = lane * 2 + j;                 // 64 float4 per 256-wide row
      v4f cv = cbr[c];
      v4f zv = zr4[c];
      q4[c] = cv;
      v4f d = cv - zv;
      le += d[0]*d[0] + d[1]*d[1] + d[2]*d[2] + d[3]*d[3];
    }
    for (int off = 16; off; off >>= 1) le += __shfl_xor(le, off);
    if (lane == 0) atomicAdd(lAcc, le);
  }
  __syncthreads();

  for (int k = t; k < KLAT; k += 256) atomicAdd(&probs_acc[k], pAcc[k]);
  if (t == 0) atomicAdd(loss_acc, lAcc[0]);
}

// ---------------- finalize: probs clip + loss scale ----------------
__global__ __launch_bounds__(1024) void vq_finish(const float* __restrict__ probs_acc,
                                                  const float* __restrict__ loss_acc,
                                                  float* __restrict__ probs_out,
                                                  float* __restrict__ loss_out) {
  int t = threadIdx.x;
  if (t < KLAT) {
    float p = probs_acc[t] * (1.0f / (float)N_TOKENS);
    probs_out[t] = fminf(fmaxf(p, 0.001f), 0.999f);
  }
  // vq_loss = (1 + BETA) * K * sum(diff^2) / (N*D)
  if (t == 0) loss_out[0] = loss_acc[0] * (1.25f * 1024.f / (65536.f * 256.f));
}

extern "C" void kernel_launch(void* const* d_in, const int* in_sizes, int n_in,
                              void* d_out, int out_size, void* d_ws, size_t ws_size,
                              hipStream_t stream) {
  (void)in_sizes; (void)n_in; (void)out_size; (void)ws_size;
  const float* z  = (const float*)d_in[0];   // latent_embedding [65536,256]
  const float* cb = (const float*)d_in[1];   // codebook [1024,256]

  float* qout      = (float*)d_out;                       // [65536,256]
  float* cb_out    = qout + (size_t)N_TOKENS * DIM;       // [1024,256]
  float* probs_out = cb_out + (size_t)KLAT * DIM;         // [1024]
  float* loss_out  = probs_out + KLAT;                    // [1]

  char*   ws        = (char*)d_ws;
  bf16_t* cb_bf     = (bf16_t*)ws;                        // 512 KB
  float*  cnorm     = (float*)(ws + 524288);              // 4 KB
  float*  probs_acc = (float*)(ws + 524288 + 4096);       // 4 KB
  float*  loss_acc  = (float*)(ws + 524288 + 8192);       // 4 B

  // codebook passthrough output
  hipMemcpyAsync(cb_out, cb, sizeof(float) * KLAT * DIM, hipMemcpyDeviceToDevice, stream);

  vq_prep<<<KLAT, 256, 0, stream>>>(cb, cb_bf, cnorm, probs_acc, loss_acc);

  size_t smem = (size_t)(MT + 2 * NCHUNK) * LDA * 2 + (size_t)MT * LDD * 4 + KLAT * 4 + 16;
  hipFuncSetAttribute((const void*)vq_main, hipFuncAttributeMaxDynamicSharedMemorySize, (int)smem);
  vq_main<<<N_TOKENS / MT, 256, smem, stream>>>(z, cb, cb_bf, cnorm, qout, probs_acc, loss_acc);

  vq_finish<<<1, 1024, 0, stream>>>(probs_acc, loss_acc, probs_out, loss_out);
}
